// ModernBertAttention_3822520894040
// MI455X (gfx1250) — matrix operand order repaired and checked by reading, not compiled, not run on hardware
//
#include <hip/hip_runtime.h>

// ---------------------------------------------------------------------------
// ModernBERT attention block for MI455X (gfx1250, wave32, WMMA).
// GEMMs on v_wmma_f32_16x16x32_bf16 (fp32 accum). Global->LDS staging uses
// CDNA5 async-to-LDS DMA (GLOBAL_LOAD_ASYNC_TO_LDS_B128 + s_wait_asynccnt)
// with double-buffered LDS so DMA for tile i+1 overlaps WMMA on tile i.
// ---------------------------------------------------------------------------

typedef unsigned short u16;
typedef __attribute__((ext_vector_type(16))) __bf16 v16bf;
typedef __attribute__((ext_vector_type(8)))  float  v8f;

constexpr int kB = 4, kS = 2048, kHid = 1024, kH = 16, kD = 64;
constexpr int kM = kB * kS; // 8192 tokens

__device__ __forceinline__ u16 f32_to_bf16(float f) {
  union { float f; unsigned u; } a; a.f = f;
  unsigned u = a.u;
  u += 0x7FFFu + ((u >> 16) & 1u); // round-to-nearest-even
  return (u16)(u >> 16);
}
__device__ __forceinline__ float bf16_to_f32(u16 h) {
  union { unsigned u; float f; } a; a.u = ((unsigned)h) << 16; return a.f;
}

union Frag { v16bf v; uint4 q[2]; };

__device__ __forceinline__ v8f wmma_bf16(const Frag& a, const Frag& b, v8f c) {
  return __builtin_amdgcn_wmma_f32_16x16x32_bf16(
      false, a.v, false, b.v, (short)0, c, false, false);
}

// CDNA5 async global->LDS DMA (per-lane 16B), tracked by ASYNCcnt.
// LDS dst VGPR takes the wave-relative LDS address = low 32 bits of the
// generic pointer to __shared__ (flat LDS aperture maps addr[31:0] to LDS).
__device__ __forceinline__ void async_load_b128(void* lds_dst, const void* gsrc) {
  unsigned ldsoff = (unsigned)(size_t)lds_dst;
  asm volatile("global_load_async_to_lds_b128 %0, %1, off"
               :: "v"(ldsoff), "v"(gsrc) : "memory");
}
__device__ __forceinline__ void wait_async0() {
  asm volatile("s_wait_asynccnt 0x0" ::: "memory");
}

// ---------------------------------------------------------------------------
// fp32 -> bf16 bulk conversion, 4 elements / thread
// ---------------------------------------------------------------------------
__global__ void cvt4_kernel(const float* __restrict__ in, u16* __restrict__ out, int n4) {
  int i = blockIdx.x * 256 + threadIdx.x;
  if (i >= n4) return;
  float4 v = ((const float4*)in)[i];
  union { u16 h[4]; uint2 u; } t;
  t.h[0] = f32_to_bf16(v.x); t.h[1] = f32_to_bf16(v.y);
  t.h[2] = f32_to_bf16(v.z); t.h[3] = f32_to_bf16(v.w);
  ((uint2*)out)[i] = t.u;
}

// ---------------------------------------------------------------------------
// LDS-tiled bf16 WMMA GEMM:  C[M,N] = A[M,K] @ B[K,N] + bias[N]
// Block tile 128x64, 8 waves (4x2), each wave a 32x32 tile (2x2 WMMAs).
// A staged via async DMA; B staged transposed via VGPRs. Double-buffered.
// ---------------------------------------------------------------------------
template <bool OUT_F32>
__global__ __launch_bounds__(256) void gemm_kernel(
    const u16* __restrict__ A, const u16* __restrict__ Bm,
    const float* __restrict__ bias, void* __restrict__ Cout,
    int M, int N, int K)
{
  __shared__ u16 lA[2][128][32];  // 16 KB
  __shared__ u16 lBT[2][64][32];  //  8 KB

  const int tid  = threadIdx.x;
  const int lane = tid & 31;
  const int w    = tid >> 5;
  const int wm   = w & 3, wn = w >> 2;
  const int half = lane >> 4, l15 = lane & 15;
  const int bm   = blockIdx.x * 128, bn = blockIdx.y * 64;

  v8f acc[2][2] = {};

  auto stageA = [&](int bi, int k0) {
#pragma unroll
    for (int i = 0; i < 2; ++i) {
      int c = tid + i * 256;
      int row = c >> 2, cg = (c & 3) * 8;
      async_load_b128(&lA[bi][row][cg], &A[(size_t)(bm + row) * K + k0 + cg]);
    }
  };
  auto stageB = [&](int bi, int k0) {
#pragma unroll
    for (int i = 0; i < 2; ++i) {
      int c = tid + i * 256;
      int r = c >> 4, cc = (c & 15) * 4;
      union { uint2 u; u16 h[4]; } t;
      t.u = *(const uint2*)&Bm[(size_t)(k0 + r) * N + bn + cc];
#pragma unroll
      for (int j = 0; j < 4; ++j) lBT[bi][cc + j][r] = t.h[j];
    }
  };

  stageA(0, 0);
  stageB(0, 0);

  for (int k0 = 0, it = 0; k0 < K; k0 += 32, ++it) {
    const int bi = it & 1;
    wait_async0();        // our wave's async A-tile for buffer bi landed
    __syncthreads();      // everyone's stores/DMA for bi visible; prev compute done
    if (k0 + 32 < K) { stageA(bi ^ 1, k0 + 32); stageB(bi ^ 1, k0 + 32); }

    Frag af[2], bf[2];
#pragma unroll
    for (int mi = 0; mi < 2; ++mi) {
      // A 16x32: lanes0-15 row=l15 K=0..15; lanes16-31 K=16..31
      const u16* p = &lA[bi][wm * 32 + mi * 16 + l15][half * 16];
      af[mi].q[0] = *(const uint4*)p;
      af[mi].q[1] = *(const uint4*)(p + 8);
    }
#pragma unroll
    for (int ni = 0; ni < 2; ++ni) {
      // B 32x16: N=l15, lanes0-15 K=0..15, lanes16-31 K=16..31
      const u16* p = &lBT[bi][wn * 32 + ni * 16 + l15][half * 16];
      bf[ni].q[0] = *(const uint4*)p;
      bf[ni].q[1] = *(const uint4*)(p + 8);
    }
#pragma unroll
    for (int mi = 0; mi < 2; ++mi)
#pragma unroll
      for (int ni = 0; ni < 2; ++ni)
        acc[mi][ni] = wmma_bf16(af[mi], bf[ni], acc[mi][ni]);
  }

  // epilogue: C/D layout -> VGPR r holds row r + 8*half, col = l15
  u16*   Cb = (u16*)Cout;
  float* Cf = (float*)Cout;
#pragma unroll
  for (int mi = 0; mi < 2; ++mi)
#pragma unroll
    for (int ni = 0; ni < 2; ++ni) {
      int col = bn + wn * 32 + ni * 16 + l15;
      float bv = bias[col];
#pragma unroll
      for (int r = 0; r < 8; ++r) {
        int mrow = bm + wm * 32 + mi * 16 + half * 8 + r;
        float v = acc[mi][ni][r] + bv;
        if constexpr (OUT_F32) Cf[(size_t)mrow * N + col] = v;
        else                   Cb[(size_t)mrow * N + col] = f32_to_bf16(v);
      }
    }
}

// ---------------------------------------------------------------------------
// RoPE + repack: qkv[B,S,3,H,D] -> q,k in [B,H,S,D] (rotated), v^T in [B,H,D,S]
// ---------------------------------------------------------------------------
__global__ void rope_kernel(const u16* __restrict__ qkv,
                            u16* __restrict__ qr, u16* __restrict__ kr,
                            u16* __restrict__ vt)
{
  int idx = blockIdx.x * 256 + threadIdx.x;
  if (idx >= kB * kS * kH * kD) return;
  int d = idx & (kD - 1);
  int h = (idx >> 6) & (kH - 1);
  int s = (idx >> 10) & (kS - 1);
  int b = idx >> 21;

  const u16* base = qkv + ((size_t)(b * kS + s)) * (3 * kHid) + h * kD;
  int j = d & 31;
  // inv_freq = 10000^(-j/32) = exp(-j * ln(10000)/32)
  float theta = (float)s * __expf(-(float)j * (9.210340371976184f / 32.0f));
  float sn, cs;
  __sincosf(theta, &sn, &cs);

  size_t oidx = (((size_t)(b * kH + h)) * kS + s) * kD + d;
  { // q
    float xv = bf16_to_f32(base[d]);
    float xr = (d < 32) ? -bf16_to_f32(base[d + 32]) : bf16_to_f32(base[d - 32]);
    qr[oidx] = f32_to_bf16(xv * cs + xr * sn);
  }
  { // k
    float xv = bf16_to_f32(base[kHid + d]);
    float xr = (d < 32) ? -bf16_to_f32(base[kHid + d + 32]) : bf16_to_f32(base[kHid + d - 32]);
    kr[oidx] = f32_to_bf16(xv * cs + xr * sn);
  }
  vt[(((size_t)(b * kH + h)) * kD + d) * kS + s] = base[2 * kHid + d];
}

// ---------------------------------------------------------------------------
// Flash attention: block per (b,h,128-query tile). 8 waves x 16 rows.
// K/V^T tiles double-buffered in LDS, filled by async DMA overlapping WMMA.
// ---------------------------------------------------------------------------
__global__ __launch_bounds__(256) void attn_kernel(
    const u16* __restrict__ qr, const u16* __restrict__ kr,
    const u16* __restrict__ vt, const int* __restrict__ pmask,
    u16* __restrict__ ctx)
{
  __shared__ u16 lK[2][128][64];   // [buf][key][d]    32 KB
  __shared__ u16 lV[2][64][128];   // [buf][d][key]    32 KB
  __shared__ u16 lP[128][128];     // [qrow][key]      32 KB

  const int tid  = threadIdx.x;
  const int lane = tid & 31, w = tid >> 5;
  const int half = lane >> 4, l15 = lane & 15;
  const int bh = blockIdx.y;
  const int b  = bh >> 4, h = bh & 15;
  const int qb = blockIdx.x * 128;

  const u16* Qb = qr + ((size_t)bh * kS + qb) * kD;
  const u16* Kb = kr + (size_t)bh * kS * kD;
  const u16* Vb = vt + (size_t)bh * kD * kS;

  auto stageKV = [&](int bi, int kt) {
#pragma unroll
    for (int i = 0; i < 4; ++i) {
      int c = tid + i * 256;
      int row = c >> 3, cg = (c & 7) * 8;
      async_load_b128(&lK[bi][row][cg], &Kb[(size_t)(kt + row) * kD + cg]);
    }
#pragma unroll
    for (int i = 0; i < 4; ++i) {
      int c = tid + i * 256;
      int row = c >> 4, cg = (c & 15) * 8;
      async_load_b128(&lV[bi][row][cg], &Vb[(size_t)row * kS + kt + cg]);
    }
  };

  // Q A-fragments (16 rows x 64 d = 2 K-steps of 32), loop invariant
  Frag qf[2];
  {
    const u16* qrow = Qb + (w * 16 + l15) * kD;
#pragma unroll
    for (int s2 = 0; s2 < 2; ++s2) {
      qf[s2].q[0] = *(const uint4*)(qrow + s2 * 32 + half * 16);
      qf[s2].q[1] = *(const uint4*)(qrow + s2 * 32 + half * 16 + 8);
    }
  }

  v8f o[4] = {};
  float m[8], l[8];
#pragma unroll
  for (int r = 0; r < 8; ++r) { m[r] = -1e30f; l[r] = 0.f; }

  stageKV(0, 0);

  for (int kt = 0; kt < kS; kt += 128) {
    const int bi = (kt >> 7) & 1;
    wait_async0();        // our wave's DMA for buffer bi landed
    __syncthreads();      // all waves' DMA landed; prev-iter compute done
    if (kt + 128 < kS) stageKV(bi ^ 1, kt + 128);

    // scores: wave's 16 rows x 128 keys = 8 C-tiles, K-reduce over d=64
    v8f st[8];
#pragma unroll
    for (int nk = 0; nk < 8; ++nk) {
      v8f z = {};
#pragma unroll
      for (int s2 = 0; s2 < 2; ++s2) {
        Frag bfK;
        const u16* p = &lK[bi][nk * 16 + l15][s2 * 32 + half * 16];
        bfK.q[0] = *(const uint4*)p;
        bfK.q[1] = *(const uint4*)(p + 8);
        z = wmma_bf16(qf[s2], bfK, z);
      }
      st[nk] = z;
    }

    // scale (1/sqrt(64)) + padding mask
#pragma unroll
    for (int nk = 0; nk < 8; ++nk) {
      int key = kt + nk * 16 + l15;
      float madd = pmask[b * kS + key] ? 0.f : -1e9f;
#pragma unroll
      for (int r = 0; r < 8; ++r) st[nk][r] = st[nk][r] * 0.125f + madd;
    }

    // online softmax: rows live in 16-lane halves (C layout)
    float ps[8];
#pragma unroll
    for (int r = 0; r < 8; ++r) {
      float mx = st[0][r];
#pragma unroll
      for (int nk = 1; nk < 8; ++nk) mx = fmaxf(mx, st[nk][r]);
#pragma unroll
      for (int i = 1; i < 16; i <<= 1) mx = fmaxf(mx, __shfl_xor(mx, i, 32));
      float mn = fmaxf(m[r], mx);
      float c  = __expf(m[r] - mn);
      l[r] *= c;
#pragma unroll
      for (int nd = 0; nd < 4; ++nd) o[nd][r] *= c;
      m[r]  = mn;
      ps[r] = 0.f;
    }
#pragma unroll
    for (int nk = 0; nk < 8; ++nk) {
#pragma unroll
      for (int r = 0; r < 8; ++r) {
        float p = __expf(st[nk][r] - m[r]);
        ps[r] += p;
        lP[w * 16 + half * 8 + r][nk * 16 + l15] = f32_to_bf16(p);
      }
    }
#pragma unroll
    for (int r = 0; r < 8; ++r) {
      float sum = ps[r];
#pragma unroll
      for (int i = 1; i < 16; i <<= 1) sum += __shfl_xor(sum, i, 32);
      l[r] += sum;
    }

    // O += P @ V : K-reduce over 128 keys (4 steps of 32), N = d (4 tiles).
    // lP is written and read by the same wave; per-wave LDS ordering + the
    // compiler's dscnt waits make this safe without a block barrier.
#pragma unroll
    for (int ks = 0; ks < 4; ++ks) {
      Frag pf;
      const u16* pp = &lP[w * 16 + l15][ks * 32 + half * 16];
      pf.q[0] = *(const uint4*)pp;
      pf.q[1] = *(const uint4*)(pp + 8);
#pragma unroll
      for (int nd = 0; nd < 4; ++nd) {
        Frag vf;
        const u16* vp = &lV[bi][nd * 16 + l15][ks * 32 + half * 16];
        vf.q[0] = *(const uint4*)vp;
        vf.q[1] = *(const uint4*)(vp + 8);
        o[nd] = wmma_bf16(pf, vf, o[nd]);
      }
    }
  }

  // normalize and write context as [B,S,HID] bf16 (row-major M for out-proj)
#pragma unroll
  for (int r = 0; r < 8; ++r) {
    float inv = 1.f / l[r];
    int row = qb + w * 16 + half * 8 + r;
#pragma unroll
    for (int nd = 0; nd < 4; ++nd) {
      int dcol = nd * 16 + l15;
      ctx[((size_t)(b * kS + row)) * kHid + h * kD + dcol] =
          f32_to_bf16(o[nd][r] * inv);
    }
  }
}

// ---------------------------------------------------------------------------
// Launcher
// ---------------------------------------------------------------------------
extern "C" void kernel_launch(void* const* d_in, const int* in_sizes, int n_in,
                              void* d_out, int out_size, void* d_ws, size_t ws_size,
                              hipStream_t stream)
{
  const float* x    = (const float*)d_in[0];
  const int*   pm   = (const int*)d_in[1];
  const float* Wqkv = (const float*)d_in[2];
  const float* bqkv = (const float*)d_in[3];
  const float* Wo   = (const float*)d_in[4];
  const float* bo   = (const float*)d_in[5];
  float* out = (float*)d_out;

  // workspace layout (bf16 as u16), ~143 MB total
  u16* xb    = (u16*)d_ws;                                // 8192*1024
  u16* wqkvb = xb    + (size_t)kM * kHid;                 // 1024*3072
  u16* wob   = wqkvb + (size_t)kHid * 3 * kHid;           // 1024*1024
  u16* qkv   = wob   + (size_t)kHid * kHid;               // 8192*3072
  u16* qr    = qkv   + (size_t)kM * 3 * kHid;             // [B,H,S,D]
  u16* kr    = qr    + (size_t)kM * kHid;                 // [B,H,S,D]
  u16* vt    = kr    + (size_t)kM * kHid;                 // [B,H,D,S]
  u16* ctx   = vt    + (size_t)kM * kHid;                 // [B,S,HID]

  int n;
  n = kM * kHid;        cvt4_kernel<<<n / 4 / 256, 256, 0, stream>>>(x,    xb,    n / 4);
  n = kHid * 3 * kHid;  cvt4_kernel<<<n / 4 / 256, 256, 0, stream>>>(Wqkv, wqkvb, n / 4);
  n = kHid * kHid;      cvt4_kernel<<<n / 4 / 256, 256, 0, stream>>>(Wo,   wob,   n / 4);

  // QKV projection: [8192,1024] x [1024,3072] -> bf16 qkv
  gemm_kernel<false><<<dim3(kM / 128, 3 * kHid / 64), 256, 0, stream>>>(
      xb, wqkvb, bqkv, (void*)qkv, kM, 3 * kHid, kHid);

  // RoPE + repack
  rope_kernel<<<(kM * kHid) / 256, 256, 0, stream>>>(qkv, qr, kr, vt);

  // attention
  attn_kernel<<<dim3(kS / 128, kB * kH), 256, 0, stream>>>(qr, kr, vt, pm, ctx);

  // output projection: [8192,1024] x [1024,1024] -> fp32 out
  gemm_kernel<true><<<dim3(kM / 128, kHid / 64), 256, 0, stream>>>(
      ctx, wob, bo, (void*)out, kM, kHid, kHid);
}
